// GraphAutoEncoder_66597762892108
// MI455X (gfx1250) — matrix operand order repaired and checked by reading, not compiled, not run on hardware
//
#include <hip/hip_runtime.h>

// ---------------------------------------------------------------------------
// GCN autoencoder on MI455X (gfx1250, wave32, WMMA).
//   deg/norm -> scale+cvt bf16 -> WMMA GEMM -> edge scatter (f32 atomics)
//   -> norm(+relu)+cvt -> WMMA GEMM -> scatter -> norm+cvt -> z z^T via WMMA
// All GEMMs use v_wmma_f32_16x16x32_bf16 (bf16 in, f32 accumulate).
// ---------------------------------------------------------------------------

typedef __bf16 bf16;
typedef __attribute__((ext_vector_type(8)))  __bf16 v8bf;
typedef __attribute__((ext_vector_type(16))) __bf16 v16bf;
typedef __attribute__((ext_vector_type(8)))  float  v8f;

#define GCN_N     16384
#define GCN_E     524288
#define GCN_INF   1024
#define GCN_HIDF  512
#define GCN_OUTF  128

// -------------------------------- utility kernels --------------------------

__global__ void k_zero(float* __restrict__ p, long long n) {
    long long i = (long long)blockIdx.x * blockDim.x + threadIdx.x;
    long long stride = (long long)gridDim.x * blockDim.x;
    for (; i < n; i += stride) p[i] = 0.0f;
}

__global__ void k_degree(const int* __restrict__ src, float* __restrict__ deg, int E) {
    int e = blockIdx.x * blockDim.x + threadIdx.x;
    if (e < E) atomicAdd(&deg[src[e]], 1.0f);
}

__global__ void k_norm(float* __restrict__ nrm, int n) {
    int i = blockIdx.x * blockDim.x + threadIdx.x;
    if (i < n) {
        float d = nrm[i];
        nrm[i] = rsqrtf(d > 0.0f ? d : 1.0f);
    }
}

// out_bf16[i] = (relu?)( x[i] * norm[i / F] ), row-major x of width F
__global__ void k_post(const float* __restrict__ x, const float* __restrict__ nrm,
                       bf16* __restrict__ out, int F, int total, int do_relu) {
    int i = blockIdx.x * blockDim.x + threadIdx.x;
    int stride = gridDim.x * blockDim.x;
    for (; i < total; i += stride) {
        float v = x[i] * nrm[i / F];
        if (do_relu) v = v > 0.0f ? v : 0.0f;
        out[i] = (bf16)v;
    }
}

// Wt[n*K + k] = bf16( W[k*Fn + n] )  (transpose so B-columns are contiguous)
__global__ void k_cvt_wt(const float* __restrict__ W, bf16* __restrict__ Wt,
                         int K, int Fn) {
    int i = blockIdx.x * blockDim.x + threadIdx.x;
    int stride = gridDim.x * blockDim.x;
    int total = K * Fn;
    for (; i < total; i += stride) {
        int k = i / Fn;
        int n = i - k * Fn;
        Wt[n * K + k] = (bf16)W[i];
    }
}

// u[src[e]*F + f] += t[dst[e]*F + f]   (grid = E blocks, block = F threads)
__global__ void k_scatter(const float* __restrict__ t, const int* __restrict__ src,
                          const int* __restrict__ dst, float* __restrict__ u, int F) {
    int e = blockIdx.x;
    int f = threadIdx.x;
    int s = src[e];
    int d = dst[e];
    atomicAdd(&u[s * F + f], t[d * F + f]);
}

// -------------------------- WMMA GEMM: C = A * Bt^T -------------------------
// A : M  x K  bf16 row-major
// Bt : Nc x K  bf16 row-major (i.e. B transposed; column n of B = row n of Bt)
// C : M  x Nc f32 row-major
// Each wave computes a 32x16 tile of C (two 16x16 accumulators sharing one
// B fragment). 8 waves/block -> 32 rows x 128 cols per block.
// grid = dim3(Nc/128, M/32), block = 256.
template <bool NT>
__global__ __launch_bounds__(256)
void k_gemm_abt(const bf16* __restrict__ A, const bf16* __restrict__ Bt,
                float* __restrict__ C, int M, int Nc, int K) {
    const int lane = threadIdx.x & 31;
    const int wave = threadIdx.x >> 5;
    const int i0 = blockIdx.y * 32;
    const int j0 = blockIdx.x * 128 + wave * 16;

    // wave32 16-bit A fragment (16x32): lanes 0-15 row m, K in [sel, sel+8) and
    // [16+sel, 16+sel+8); lanes 16-31 same rows, sel = 8.
    const int rowA = lane & 15;
    const int selA = (lane >> 4) * 8;
    // wave32 16-bit B fragment (32x16): lane covers column n = lane&15, K in
    // [sel, sel+8) and [16+sel, 16+sel+8), sel = 8 for lanes 16-31 (matches A).
    const int colB = lane & 15;
    const int selB = (lane >> 4) * 8;

    const bf16* pa0 = A + (size_t)(i0 + rowA) * K;
    const bf16* pa1 = pa0 + (size_t)16 * K;
    const bf16* pb  = Bt + (size_t)(j0 + colB) * K;

    v8f acc0 = {};
    v8f acc1 = {};

    for (int k = 0; k < K; k += 32) {
        v8bf a0lo = *(const v8bf*)(pa0 + k + selA);
        v8bf a0hi = *(const v8bf*)(pa0 + k + selA + 16);
        v8bf a1lo = *(const v8bf*)(pa1 + k + selA);
        v8bf a1hi = *(const v8bf*)(pa1 + k + selA + 16);
        v8bf blo  = *(const v8bf*)(pb  + k + selB);
        v8bf bhi  = *(const v8bf*)(pb  + k + selB + 16);

        v16bf av0, av1, bv;
#pragma unroll
        for (int t = 0; t < 8; ++t) {
            av0[t] = a0lo[t]; av0[t + 8] = a0hi[t];
            av1[t] = a1lo[t]; av1[t + 8] = a1hi[t];
            bv[t]  = blo[t];  bv[t + 8]  = bhi[t];
        }
        // (neg_a, A, neg_b, B, c_mod, C, reuse_a, reuse_b)
        acc0 = __builtin_amdgcn_wmma_f32_16x16x32_bf16(false, av0, false, bv,
                                                       (short)0, acc0, false, false);
        acc1 = __builtin_amdgcn_wmma_f32_16x16x32_bf16(false, av1, false, bv,
                                                       (short)0, acc1, false, false);
    }

    // C fragment layout: lanes 0-15 -> M = r, lanes 16-31 -> M = 8 + r; N = lane&15
    const int n  = lane & 15;
    const int mb = (lane >> 4) * 8;
    float* pc0 = C + (size_t)(i0 + mb) * Nc + j0 + n;
    float* pc1 = pc0 + (size_t)16 * Nc;
#pragma unroll
    for (int r = 0; r < 8; ++r) {
        if (NT) {
            __builtin_nontemporal_store(acc0[r], pc0 + (size_t)r * Nc);
            __builtin_nontemporal_store(acc1[r], pc1 + (size_t)r * Nc);
        } else {
            pc0[(size_t)r * Nc] = acc0[r];
            pc1[(size_t)r * Nc] = acc1[r];
        }
    }
}

// ------------------------------- launcher ----------------------------------

extern "C" void kernel_launch(void* const* d_in, const int* in_sizes, int n_in,
                              void* d_out, int out_size, void* d_ws, size_t ws_size,
                              hipStream_t stream) {
    (void)in_sizes; (void)n_in; (void)out_size; (void)ws_size;

    const float* h   = (const float*)d_in[0];
    const int*   src = (const int*)d_in[1];
    const int*   dst = (const int*)d_in[2];
    const float* W0  = (const float*)d_in[3];
    const float* W1  = (const float*)d_in[4];
    float* out = (float*)d_out;

    const int N = GCN_N, E = GCN_E, IN_F = GCN_INF, HID_F = GCN_HIDF, OUT_F = GCN_OUTF;

    // ---- workspace layout (all offsets 256B aligned) ----
    char* w = (char*)d_ws;
    float* nrm = (float*)w;                                  // N f32        (64 KB)
    size_t o = (size_t)N * sizeof(float);
    bf16* W0t = (bf16*)(w + o); o += (size_t)IN_F * HID_F * 2;   // 1 MB
    bf16* W1t = (bf16*)(w + o); o += (size_t)HID_F * OUT_F * 2;  // 128 KB
    char* regA = w + o; o += (size_t)N * IN_F * 2;               // 32 MB, multi-use
    char* regB = w + o;                                          // 32 MB, multi-use

    bf16*  hs = (bf16*)regA;                                 // N x IN_F bf16
    float* t1 = (float*)regB;                                // N x HID_F f32
    float* u1 = (float*)regA;                                // N x HID_F f32 (hs dead)
    bf16*  h1 = (bf16*)regB;                                 // N x HID_F bf16 (t1 dead)
    float* t2 = (float*)regA;                                // N x OUT_F f32 (u1 dead)
    float* u2 = (float*)(regA + (size_t)N * OUT_F * 4);      // N x OUT_F f32
    bf16*  z  = (bf16*)(regA + (size_t)2 * N * OUT_F * 4);   // N x OUT_F bf16

    // ---- 1) degree + symmetric norm ----
    k_zero<<<256, 256, 0, stream>>>(nrm, (long long)N);
    k_degree<<<(E + 255) / 256, 256, 0, stream>>>(src, nrm, E);
    k_norm<<<(N + 255) / 256, 256, 0, stream>>>(nrm, N);

    // ---- 2) weight conversion (transposed, bf16) ----
    k_cvt_wt<<<1024, 256, 0, stream>>>(W0, W0t, IN_F, HID_F);
    k_cvt_wt<<<256, 256, 0, stream>>>(W1, W1t, HID_F, OUT_F);

    // ---- 3) layer 1: hs = bf16(h * norm); t1 = hs @ W0; u1 = scatter; h1 ----
    k_post<<<4096, 256, 0, stream>>>(h, nrm, hs, IN_F, N * IN_F, 0);
    k_gemm_abt<false><<<dim3(HID_F / 128, N / 32), 256, 0, stream>>>(hs, W0t, t1, N, HID_F, IN_F);
    k_zero<<<4096, 256, 0, stream>>>(u1, (long long)N * HID_F);
    k_scatter<<<E, HID_F, 0, stream>>>(t1, src, dst, u1, HID_F);
    k_post<<<4096, 256, 0, stream>>>(u1, nrm, h1, HID_F, N * HID_F, 1);

    // ---- 4) layer 2: t2 = h1 @ W1; u2 = scatter; z = bf16(u2 * norm) ----
    k_gemm_abt<false><<<dim3(OUT_F / 128, N / 32), 256, 0, stream>>>(h1, W1t, t2, N, OUT_F, HID_F);
    k_zero<<<1024, 256, 0, stream>>>(u2, (long long)N * OUT_F);
    k_scatter<<<E, OUT_F, 0, stream>>>(t2, src, dst, u2, OUT_F);
    k_post<<<1024, 256, 0, stream>>>(u2, nrm, z, OUT_F, N * OUT_F, 0);

    // ---- 5) decoder: a_hat = z @ z^T  (1 GiB streaming output, NT stores) ----
    k_gemm_abt<true><<<dim3(N / 128, N / 32), 256, 0, stream>>>(z, z, out, N, N, OUT_F);
}
